// GroupedExperts_2233382993921
// MI455X (gfx1250) — matrix operand order repaired and checked by reading, not compiled, not run on hardware
//
#include <hip/hip_runtime.h>

// ---------------------------------------------------------------------------
// Grouped experts MLP: h = x@W1 + b1 ; h' = swiglu(h) ; out = h'@W2 + b2
// E=8, T=2048, D=2048, N1=2*D=4096.  Compute-bound (~0.41 TFLOP vs ~0.6 GB),
// so both GEMMs run through v_wmma_f32_16x16x32_bf16 (f32 accumulate).
//  - f32->bf16 staging via hardware v_cvt_pk_bf16_f32 (1 VALU / 2 elems)
//  - single barrier per K-step double-buffer pipeline
//  - GEMM2 A-tile (bf16) staged with global_load_async_to_lds_b128 (ASYNCcnt)
//  - bias pre-loaded into WMMA accumulators
// ---------------------------------------------------------------------------

typedef __bf16 bf16;
typedef __attribute__((ext_vector_type(16))) __bf16 v16bf;
typedef __attribute__((ext_vector_type(8)))  __bf16 v8bf;
typedef __attribute__((ext_vector_type(8)))  float  v8f;

static constexpr int E  = 8;
static constexpr int T  = 2048;
static constexpr int D  = 2048;
static constexpr int N1 = 2 * D;      // 4096
static constexpr int KT = D / 32;     // 64 K-steps of 32
static constexpr int LDT = 40;        // LDS tile row stride in bf16 (80B, 16B aligned)
static constexpr float ALPHA = 1.702f;
static constexpr float LIMIT = 7.0f;

// Packed f32x2 -> bf16x2 (RNE) in one VALU op.
__device__ __forceinline__ unsigned cvt_pk_bf16(float lo, float hi) {
  unsigned r;
  asm("v_cvt_pk_bf16_f32 %0, %1, %2" : "=v"(r) : "v"(lo), "v"(hi));
  return r;
}

__device__ __forceinline__ v8f splat8(float x) {
  v8f v = {x, x, x, x, x, x, x, x};
  return v;
}

// Async global->LDS copy of 16B per lane (GV mode, ASYNCcnt-tracked).
// Generic pointers to LDS carry the LDS offset in their low 32 bits.
__device__ __forceinline__ void async_ld_b128(void* lds_ptr, const void* g) {
  unsigned loff = (unsigned)(uintptr_t)lds_ptr;
  asm volatile("global_load_async_to_lds_b128 %0, %1, off"
               :: "v"(loff), "v"(g) : "memory");
}
__device__ __forceinline__ void wait_async0() {
  asm volatile("s_wait_asynccnt 0x0" ::: "memory");
}

__device__ __forceinline__ v16bf cat8(v8bf lo, v8bf hi) {
  return __builtin_shufflevector(lo, hi, 0,1,2,3,4,5,6,7,8,9,10,11,12,13,14,15);
}

// A fragment (16x32 bf16): lane l<16 -> row l, K in [8h,8h+8) then [16+8h,16+8h+8)
__device__ __forceinline__ v16bf load_a_frag(const bf16* rowp, int kh) {
  v8bf lo = *(const v8bf*)(rowp + kh * 8);
  v8bf hi = *(const v8bf*)(rowp + 16 + kh * 8);
  return cat8(lo, hi);
}
// B fragment (32x16 bf16): lane l -> column l&15, K in [8h,8h+8) then [16+8h,16+8h+8)
__device__ __forceinline__ v16bf load_b_frag(const bf16* rowp, int kh) {
  v8bf lo = *(const v8bf*)(rowp + kh * 8);
  v8bf hi = *(const v8bf*)(rowp + 16 + kh * 8);
  return cat8(lo, hi);
}

// ======================= Kernel 1: GEMM1 + bias + SwiGLU ====================
__global__ __launch_bounds__(256)
void gemm1_swiglu_kernel(const float* __restrict__ x,
                         const float* __restrict__ w1,
                         const float* __restrict__ b1,
                         bf16* __restrict__ hws) {
  __shared__ bf16 sA[2][128 * LDT];
  __shared__ bf16 sB[2][128 * LDT];
  __shared__ float sH[128 * 132];

  const int tid  = threadIdx.x;
  const int lane = tid & 31;
  const int wid  = tid >> 5;
  const int wm   = wid >> 2;          // 0..1  (64-row band)
  const int wn   = wid & 3;           // 0..3  (32-col band)
  const int kh   = lane >> 4;
  const int l15  = lane & 15;

  const int e  = blockIdx.z;
  const int m0 = blockIdx.y * 128;    // rows of h (T)
  const int n0 = blockIdx.x * 128;    // cols of h (N1)

  const float* Ag = x  + ((size_t)e * T + m0) * D;          // ld = D
  const float* Bg = w1 + (size_t)e * D * N1 + n0;           // ld = N1
  const float* b1e = b1 + (size_t)e * N1 + n0;

  // A staging: row = ar + 32q, 8 float4 segs per row
  const int ar = tid >> 3, as = tid & 7;
  // B staging: column pair bn2, 8 float2 along K
  const int bn2 = (tid & 63) * 2, bh = tid >> 6;   // bh in 0..3 -> K octet

  float4 areg[4];
  float2 breg[8];
  v8f    acc[4][2];
  // Fold bias into accumulator init (column-only, shared by all 8 C VGPRs).
#pragma unroll
  for (int j = 0; j < 2; ++j) {
    const float bv = b1e[wn * 32 + j * 16 + l15];
#pragma unroll
    for (int i = 0; i < 4; ++i) acc[i][j] = splat8(bv);
  }

  auto load_global = [&](int kt) {
    const int k0 = kt * 32;
#pragma unroll
    for (int q = 0; q < 4; ++q)
      areg[q] = *(const float4*)(Ag + (size_t)(ar + 32 * q) * D + k0 + as * 4);
#pragma unroll
    for (int j = 0; j < 8; ++j)
      breg[j] = *(const float2*)(Bg + (size_t)(k0 + bh * 8 + j) * N1 + bn2);
  };

  auto store_lds = [&](int buf) {
#pragma unroll
    for (int q = 0; q < 4; ++q) {
      uint2 u;
      u.x = cvt_pk_bf16(areg[q].x, areg[q].y);
      u.y = cvt_pk_bf16(areg[q].z, areg[q].w);
      *(uint2*)(&sA[buf][(ar + 32 * q) * LDT + as * 4]) = u;
    }
#pragma unroll
    for (int j = 0; j < 8; j += 2) {   // transposed: N-major, K contiguous
      *(unsigned*)(&sB[buf][bn2 * LDT + bh * 8 + j]) =
          cvt_pk_bf16(breg[j].x, breg[j + 1].x);
      *(unsigned*)(&sB[buf][(bn2 + 1) * LDT + bh * 8 + j]) =
          cvt_pk_bf16(breg[j].y, breg[j + 1].y);
    }
  };

  auto compute = [&](int buf) {
    v16bf av[4], bv[2];
#pragma unroll
    for (int i = 0; i < 4; ++i)
      av[i] = load_a_frag(&sA[buf][(wm * 64 + i * 16 + l15) * LDT], kh);
#pragma unroll
    for (int j = 0; j < 2; ++j)
      bv[j] = load_b_frag(&sB[buf][(wn * 32 + j * 16 + l15) * LDT], kh);
#pragma unroll
    for (int i = 0; i < 4; ++i)
#pragma unroll
      for (int j = 0; j < 2; ++j)
        acc[i][j] = __builtin_amdgcn_wmma_f32_16x16x32_bf16(
            false, av[i], false, bv[j], (short)0, acc[i][j], false, false);
  };

  load_global(0);
  store_lds(0);
  __syncthreads();
  // Single barrier per K-step: writes to buf `nxt` are ordered against the
  // last reads of `nxt` (previous iteration) by the barrier at its end.
  for (int kt = 0; kt < KT; ++kt) {
    const int buf = kt & 1;
    if (kt + 1 < KT) load_global(kt + 1);
    compute(buf);
    if (kt + 1 < KT) store_lds(1 - buf);
    __syncthreads();
  }

  // Stage (biased) C tile to LDS so even/odd SwiGLU columns pair locally.
#pragma unroll
  for (int i = 0; i < 4; ++i)
#pragma unroll
    for (int j = 0; j < 2; ++j)
#pragma unroll
      for (int g = 0; g < 8; ++g) {
        const int row = wm * 64 + i * 16 + kh * 8 + g;
        const int col = wn * 32 + j * 16 + l15;
        sH[row * 132 + col] = acc[i][j][g];
      }
  __syncthreads();

  // SwiGLU: each thread owns 2 adjacent output columns -> float4 LDS reads,
  // packed bf16x2 dword stores.
  unsigned* Ho = (unsigned*)((unsigned short*)hws +
                             ((size_t)e * T + m0) * D + (n0 >> 1));
  const int cp = tid & 31;          // column-pair index (out cols 2cp, 2cp+1)
  const int r0 = tid >> 5;          // 8 row phases
#pragma unroll
  for (int q = 0; q < 16; ++q) {
    const int row = r0 + q * 8;
    const float4 hq = *(const float4*)(&sH[row * 132 + 4 * cp]);
    float xg0 = fminf(hq.x, LIMIT);
    float xl0 = fminf(fmaxf(hq.y, -LIMIT), LIMIT);
    float xg1 = fminf(hq.z, LIMIT);
    float xl1 = fminf(fmaxf(hq.w, -LIMIT), LIMIT);
    const float a0 = xg0 * (1.0f / (1.0f + __expf(-ALPHA * xg0))) * (xl0 + 1.0f);
    const float a1 = xg1 * (1.0f / (1.0f + __expf(-ALPHA * xg1))) * (xl1 + 1.0f);
    Ho[((size_t)row * D + 2 * cp) >> 1] = cvt_pk_bf16(a0, a1);
  }
}

// ======================= Kernel 2: GEMM2 + bias ============================
__global__ __launch_bounds__(256)
void gemm2_kernel(const bf16* __restrict__ hws,
                  const float* __restrict__ w2,
                  const float* __restrict__ b2,
                  float* __restrict__ out) {
  __shared__ bf16 sA[2][128 * LDT];
  __shared__ bf16 sB[2][128 * LDT];

  const int tid  = threadIdx.x;
  const int lane = tid & 31;
  const int wid  = tid >> 5;
  const int wm   = wid >> 2;
  const int wn   = wid & 3;
  const int kh   = lane >> 4;
  const int l15  = lane & 15;

  const int e  = blockIdx.z;
  const int m0 = blockIdx.y * 128;
  const int n0 = blockIdx.x * 128;

  const bf16*  Ag = hws + ((size_t)e * T + m0) * D;   // bf16, ld = D
  const float* Bg = w2 + (size_t)e * D * D + n0;      // f32,  ld = D
  const float* b2e = b2 + (size_t)e * D + n0;

  const int ar2 = tid >> 1, ah2 = tid & 1;            // A: one 16-elem half-row
  const int bn2 = (tid & 63) * 2, bh = tid >> 6;      // B: column pair, 8 K

  float2 breg[8];
  v8f    acc[4][2];
#pragma unroll
  for (int j = 0; j < 2; ++j) {
    const float bv = b2e[wn * 32 + j * 16 + l15];
#pragma unroll
    for (int i = 0; i < 4; ++i) acc[i][j] = splat8(bv);
  }

  // A tile (already bf16): async global->LDS, no VGPR staging, no conversion.
  auto issue_a_async = [&](int buf, int kt) {
    const int k0 = kt * 32;
    const bf16* p = Ag + (size_t)ar2 * D + k0 + ah2 * 16;
    bf16* l = &sA[buf][ar2 * LDT + ah2 * 16];
    async_ld_b128(l, p);
    async_ld_b128(l + 8, p + 8);
  };

  auto load_b = [&](int kt) {
    const int k0 = kt * 32;
#pragma unroll
    for (int j = 0; j < 8; ++j)
      breg[j] = *(const float2*)(Bg + (size_t)(k0 + bh * 8 + j) * D + bn2);
  };

  auto store_b = [&](int buf) {
#pragma unroll
    for (int j = 0; j < 8; j += 2) {
      *(unsigned*)(&sB[buf][bn2 * LDT + bh * 8 + j]) =
          cvt_pk_bf16(breg[j].x, breg[j + 1].x);
      *(unsigned*)(&sB[buf][(bn2 + 1) * LDT + bh * 8 + j]) =
          cvt_pk_bf16(breg[j].y, breg[j + 1].y);
    }
  };

  auto compute = [&](int buf) {
    v16bf av[4], bv[2];
#pragma unroll
    for (int i = 0; i < 4; ++i)
      av[i] = load_a_frag(&sA[buf][(wm * 64 + i * 16 + l15) * LDT], kh);
#pragma unroll
    for (int j = 0; j < 2; ++j)
      bv[j] = load_b_frag(&sB[buf][(wn * 32 + j * 16 + l15) * LDT], kh);
#pragma unroll
    for (int i = 0; i < 4; ++i)
#pragma unroll
      for (int j = 0; j < 2; ++j)
        acc[i][j] = __builtin_amdgcn_wmma_f32_16x16x32_bf16(
            false, av[i], false, bv[j], (short)0, acc[i][j], false, false);
  };

  issue_a_async(0, 0);
  load_b(0);
  store_b(0);
  wait_async0();
  __syncthreads();
  for (int kt = 0; kt < KT; ++kt) {
    const int buf = kt & 1;
    if (kt + 1 < KT) { issue_a_async(1 - buf, kt + 1); load_b(kt + 1); }
    compute(buf);
    if (kt + 1 < KT) store_b(1 - buf);
    wait_async0();
    __syncthreads();
  }

  float* Og = out + ((size_t)e * T + m0) * D + n0;
#pragma unroll
  for (int i = 0; i < 4; ++i)
#pragma unroll
    for (int j = 0; j < 2; ++j)
#pragma unroll
      for (int g = 0; g < 8; ++g) {
        const int row = wm * 64 + i * 16 + kh * 8 + g;
        const int col = wn * 32 + j * 16 + l15;
        Og[(size_t)row * D + col] = acc[i][j][g];
      }
}

// ============================== launcher ===================================
extern "C" void kernel_launch(void* const* d_in, const int* in_sizes, int n_in,
                              void* d_out, int out_size, void* d_ws, size_t ws_size,
                              hipStream_t stream) {
  const float* x  = (const float*)d_in[0];
  const float* w1 = (const float*)d_in[1];
  const float* b1 = (const float*)d_in[2];
  const float* w2 = (const float*)d_in[3];
  const float* b2 = (const float*)d_in[4];
  float* out = (float*)d_out;
  bf16*  hws = (bf16*)d_ws;   // h' activations, E*T*D bf16 = 67 MB

  dim3 block(256);
  gemm1_swiglu_kernel<<<dim3(N1 / 128, T / 128, E), block, 0, stream>>>(x, w1, b1, hws);
  gemm2_kernel<<<dim3(D / 128, T / 128, E), block, 0, stream>>>(hws, w2, b2, out);
}
